// SelfAttentionV2_35940286333005
// MI455X (gfx1250) — matrix operand order repaired and checked by reading, not compiled, not run on hardware
//
#include <hip/hip_runtime.h>
#include <math.h>

// ---------------------------------------------------------------------------
// SelfAttentionV2 for MI455X (gfx1250): bf16 WMMA pipeline + TDM staging.
//   SEQ=4096, IN_DIM=1024, OUT_DIM=1024
//   Kernel 0: one-shot f32 -> bf16 conversion of X, Wq, Wk, Wv (bandwidth-bound)
//   Kernel 1: Q/K/V = X @ W^T. The shared 16x1024 A-tile is staged into LDS
//             once per workgroup by the Tensor Data Mover (tensor_load_to_lds,
//             s_wait_tensorcnt); B streams from L2; all math is bf16 WMMA.
//   Kernel 2: flash-style attention, online softmax, all matmuls via WMMA.
// ---------------------------------------------------------------------------

#define SEQ     4096
#define DIM     1024
#define KVBLK   32      // KV tile width (== WMMA K for the P@V matmul)
#define QBLK    16      // query rows per workgroup

typedef __attribute__((ext_vector_type(16))) __bf16          v16bf;
typedef __attribute__((ext_vector_type(8)))  float           v8f;
typedef __attribute__((ext_vector_type(16))) unsigned short  us16;
typedef __attribute__((ext_vector_type(8)))  unsigned short  us8;
typedef __attribute__((ext_vector_type(4)))  unsigned short  us4;
typedef __attribute__((ext_vector_type(4)))  unsigned int    v4u;
typedef __attribute__((ext_vector_type(4)))  int             v4i;
typedef __attribute__((ext_vector_type(8)))  int             v8i_;

// f32 -> bf16, round-to-nearest-even (integer path: guaranteed to lower)
static __device__ __forceinline__ unsigned short f2bf(float f) {
    union { float f; unsigned u; } v; v.f = f;
    unsigned u = v.u;
    u += 0x7FFFu + ((u >> 16) & 1u);
    return (unsigned short)(u >> 16);
}

// 16 contiguous bf16 (32B aligned) -> WMMA bf16 fragment
static __device__ __forceinline__ v16bf ld16(const unsigned short* p) {
    us16 u = *(const us16*)p;
    return __builtin_bit_cast(v16bf, u);
}

// two groups of 8 contiguous bf16 (16B aligned each) -> WMMA bf16 fragment
static __device__ __forceinline__ v16bf ld8x2(const unsigned short* p0,
                                              const unsigned short* p1) {
    us8 lo = *(const us8*)p0;
    us8 hi = *(const us8*)p1;
    us16 u;
#pragma unroll
    for (int j = 0; j < 8; ++j) { u[j] = lo[j]; u[8 + j] = hi[j]; }
    return __builtin_bit_cast(v16bf, u);
}

// ---------------------------------------------------------------------------
// TDM: stage one contiguous 16x1024 bf16 tile (32 KB) global -> LDS.
// 1D tile of 16384 2-byte elements; LDS padding: 16B after every 256 dwords
// (pad_interval=7, pad_amount=3) -> 1040B chunks, keeps ds_load_b128 aligned
// and spreads lanes over banks. D# layout per CDNA5 ISA.
// ---------------------------------------------------------------------------
#define ACHUNK 1040                         // 1024B data + 16B pad per 256 dwords
static __device__ __forceinline__ void
tdm_load_atile(const unsigned short* gsrc, unsigned lds_off) {
    const unsigned long long ga = (unsigned long long)gsrc;
    v4u  g0;
    v8i_ g1;
    v4i  gz = {0, 0, 0, 0};
    g0[0] = 1u;                                            // count=1 (valid D#)
    g0[1] = lds_off;                                       // lds_addr (bytes)
    g0[2] = (unsigned)(ga & 0xFFFFFFFFu);                  // global_addr[31:0]
    g0[3] = (unsigned)((ga >> 32) & 0x01FFFFFFu)           // global_addr[56:32]
          | (2u << 30);                                    // type = 2 ("image")
    g1[0] = (int)((1u << 16)      // data_size = 1 -> 2-byte elements
                | (1u << 20)      // pad_enable
                | (7u << 22)      // pad_interval: 256 dwords
                | (3u << 25));    // pad_amount: 4 dwords = 16B
    g1[1] = (int)((16384u & 0xFFFFu) << 16);   // tensor_dim0[15:0] -> bits 63:48
    g1[2] = (int)(1u << 16);                   // dim0[31:16]=0 | tensor_dim1=1
    g1[3] = (int)(16384u << 16);               // tile_dim0 = 16384 -> bits 127:112
    g1[4] = 0;                                 // tile_dim1 = tile_dim2 = 0
    g1[5] = 16384;                             // tensor_dim0_stride[31:0]
    g1[6] = 0;
    g1[7] = 0;
#if defined(__clang_major__) && (__clang_major__ >= 23)
    v8i_ g4 = {0, 0, 0, 0, 0, 0, 0, 0};
    __builtin_amdgcn_tensor_load_to_lds(g0, g1, gz, gz, g4, 0);
#else
    __builtin_amdgcn_tensor_load_to_lds(g0, g1, gz, gz, 0);
#endif
}

// ---------------------------------------------------------------------------
// Kernel 0: bulk f32 -> bf16 (4 elems/thread/iter, grid-stride)
// ---------------------------------------------------------------------------
__global__ void __launch_bounds__(256)
cvt_f32_bf16(const float* __restrict__ src, unsigned short* __restrict__ dst,
             int n4) {                     // n4 = element count / 4
    for (int i = blockIdx.x * blockDim.x + threadIdx.x; i < n4;
         i += gridDim.x * blockDim.x) {
        float4 f = ((const float4*)src)[i];
        us4 o;
        o[0] = f2bf(f.x); o[1] = f2bf(f.y); o[2] = f2bf(f.z); o[3] = f2bf(f.w);
        ((us4*)dst)[i] = o;
    }
}

// ---------------------------------------------------------------------------
// Kernel 1: out = X @ W^T in bf16. 8 waves share one TDM-staged A-tile in LDS;
// each wave computes 16 rows x 128 cols (8 WMMAs per K-step, A reused).
//   z==0 -> Qbf row-major [SEQ][DIM]
//   z==1 -> Kbf row-major [SEQ][DIM]
//   z==2 -> Vt  TRANSPOSED [DIM][SEQ]
// ---------------------------------------------------------------------------
__global__ void __launch_bounds__(256)
qkv_gemm_bf16(const unsigned short* __restrict__ Xbf,
              const unsigned short* __restrict__ Wqbf,
              const unsigned short* __restrict__ Wkbf,
              const unsigned short* __restrict__ Wvbf,
              unsigned short* __restrict__ Qbf,
              unsigned short* __restrict__ Kbf,
              unsigned short* __restrict__ Vt) {
    __shared__ __align__(16) unsigned char sA[16 * 2 * ACHUNK];  // padded A tile

    const int wave = threadIdx.x >> 5;
    const int lane = threadIdx.x & 31;
    const int hi   = lane >> 4;
    const int ln   = lane & 15;
    const int mt   = blockIdx.x;          // 0..255 (16-row tile)
    const int which = blockIdx.y;         // 0:Q 1:K 2:V
    const int n0   = wave * 128;          // this wave's 128-col slice

    const unsigned short* W =
        (which == 0) ? Wqbf : (which == 1) ? Wkbf : Wvbf;

    const int ab = hi * 16;               // A K-offset (0 or 16)
    const int bb = hi * 16;               // B K-offset (0 or 16)

    // ---- TDM: one wave DMAs the shared 16x1024 A tile into LDS ----------
    if (wave == 0) {
        tdm_load_atile(Xbf + (size_t)mt * 16 * DIM,
                       (unsigned)(unsigned long long)(void*)sA);
        __builtin_amdgcn_s_wait_tensorcnt(0);
    }
    __syncthreads();

    // padded-LDS addressing: row ln, column col (bf16 units)
    const unsigned char* aRow = sA + (2 * ln) * ACHUNK;
    auto aPtr = [&](int col) -> const unsigned short* {
        return (const unsigned short*)(aRow + (col >> 9) * ACHUNK
                                            + ((col & 511) << 1));
    };

    const unsigned short* wbase = W + (size_t)(n0 + ln) * DIM;

    v8f acc[8];
#pragma unroll
    for (int t = 0; t < 8; ++t) acc[t] = (v8f){};

#pragma unroll 2
    for (int k0 = 0; k0 < DIM; k0 += 32) {
        v16bf a = ld8x2(aPtr(k0 + ab), aPtr(k0 + ab + 8));    // ds_load_b128 x2
        v16bf b[8];
#pragma unroll
        for (int t = 0; t < 8; ++t)
            b[t] = ld16(wbase + (size_t)t * 16 * DIM + k0 + bb);
#pragma unroll
        for (int t = 0; t < 8; ++t)
            acc[t] = __builtin_amdgcn_wmma_f32_16x16x32_bf16(
                         false, a, false, b[t], (short)0, acc[t], false, false);
    }

    // ---- branch-free epilogue: pick (dst, row-stride, col-stride) once ----
    unsigned short* dst;
    size_t rs, cs;
    if (which == 0)      { dst = Qbf; rs = DIM; cs = 1;   }
    else if (which == 1) { dst = Kbf; rs = DIM; cs = 1;   }
    else                 { dst = Vt;  rs = 1;   cs = SEQ; }

#pragma unroll
    for (int t = 0; t < 8; ++t) {
        const size_t ncs = (size_t)(n0 + t * 16 + ln) * cs;
#pragma unroll
        for (int r = 0; r < 8; ++r) {
            const size_t row = (size_t)(mt * 16 + r + 8 * hi);
            dst[row * rs + ncs] = f2bf(acc[t][r]);
        }
    }
}

// ---------------------------------------------------------------------------
// Kernel 2: flash attention over full 1024-dim "head".
// Workgroup = 8 waves = 16 query rows. Each wave owns 128 output columns.
// Per 32-wide KV tile:
//   (1) S partials: wave w -> subtile (w&1), K-quarter (w>>1)*256
//   (2) wave 0: reduce partials, scale 1/32, online softmax -> P (bf16, LDS)
//   (3) all waves: rescale O accumulators, O += P @ V-slice (one WMMA / 16 cols)
// ---------------------------------------------------------------------------
__global__ void __launch_bounds__(256)
flash_attn_bf16(const unsigned short* __restrict__ Qbf,
                const unsigned short* __restrict__ Kbf,
                const unsigned short* __restrict__ Vt,
                float* __restrict__ Out) {
    __shared__ float          sPart[8][256];        // per-wave 16x16 S partials
    __shared__ float          sS[16][KVBLK];        // reduced, scaled scores
    __shared__ unsigned short sP[16][KVBLK];        // exp(S - m) as bf16
    __shared__ float          sC[16];               // exp(m_old - m_new)
    __shared__ float          sM[16];               // running max
    __shared__ float          sL[16];               // running denom

    const int wave = threadIdx.x >> 5;
    const int lane = threadIdx.x & 31;
    const int hi   = lane >> 4;
    const int am   = lane & 15;
    const int q0   = blockIdx.x * QBLK;

    const int sel = wave & 1;            // which 16-col S subtile
    const int kq  = (wave >> 1) * 256;   // this wave's K-reduction quarter
    const int ab  = hi * 16;             // A-frag K-offset
    const int bb  = hi * 16;             // B-frag K-offset

    if (threadIdx.x < 16) { sM[threadIdx.x] = -1e30f; sL[threadIdx.x] = 0.0f; }
    __syncthreads();

    // Preload Q fragments for this wave's K-quarter (register resident).
    v16bf qa[8];
    {
        const unsigned short* qrow = Qbf + (size_t)(q0 + am) * DIM;
#pragma unroll
        for (int s = 0; s < 8; ++s) {
            const unsigned short* p = qrow + kq + s * 32 + ab;
            qa[s] = ld8x2(p, p + 8);
        }
    }

    v8f o[8];
#pragma unroll
    for (int t = 0; t < 8; ++t) o[t] = (v8f){};

    for (int kv0 = 0; kv0 < SEQ; kv0 += KVBLK) {
        // ---- (1) partial S = Q_quarter @ K_subtile^T -------------------
        {
            const unsigned short* krow =
                Kbf + (size_t)(kv0 + sel * 16 + am) * DIM + kq + bb;
            v16bf kb[8];
#pragma unroll
            for (int s = 0; s < 8; ++s) kb[s] = ld16(krow + s * 32);
            v8f sp = {};
#pragma unroll
            for (int s = 0; s < 8; ++s)
                sp = __builtin_amdgcn_wmma_f32_16x16x32_bf16(
                         false, qa[s], false, kb[s], (short)0, sp, false, false);
#pragma unroll
            for (int r = 0; r < 8; ++r)
                sPart[wave][(r + 8 * hi) * 16 + am] = sp[r];
        }
        __syncthreads();

        // ---- (2) reduce + online softmax (wave 0, lanes 0..15 = rows) --
        if (wave == 0 && lane < 16) {
            const int row = lane;
            float rmax = -1e30f;
#pragma unroll 4
            for (int c = 0; c < KVBLK; ++c) {
                const int sub = c >> 4, cc = c & 15;
                float v = 0.0f;
#pragma unroll
                for (int qtr = 0; qtr < 4; ++qtr)
                    v += sPart[qtr * 2 + sub][row * 16 + cc];
                v *= 0.03125f;                 // 1/sqrt(1024)
                sS[row][c] = v;
                rmax = fmaxf(rmax, v);
            }
            const float mold = sM[row];
            const float mnew = fmaxf(mold, rmax);
            const float corr = __expf(mold - mnew);
            float lsum = 0.0f;
#pragma unroll 4
            for (int c = 0; c < KVBLK; ++c) {
                const float p = __expf(sS[row][c] - mnew);
                lsum += p;
                sP[row][c] = f2bf(p);
            }
            sM[row] = mnew;
            sL[row] = sL[row] * corr + lsum;
            sC[row] = corr;
        }
        __syncthreads();

        // ---- (3) rescale O, then O += P @ V-slice ----------------------
        {
            v16bf vb[8];
            const unsigned short* vbase =
                Vt + (size_t)(wave * 128 + am) * SEQ + kv0 + bb;
#pragma unroll
            for (int t = 0; t < 8; ++t)
                vb[t] = ld16(vbase + (size_t)t * 16 * SEQ);
            const unsigned short* pp = &sP[am][ab];
            v16bf pa = ld8x2(pp, pp + 8);
#pragma unroll
            for (int r = 0; r < 8; ++r) {
                const float c = sC[r + 8 * hi];
#pragma unroll
                for (int t = 0; t < 8; ++t) o[t][r] *= c;
            }
#pragma unroll
            for (int t = 0; t < 8; ++t)
                o[t] = __builtin_amdgcn_wmma_f32_16x16x32_bf16(
                           false, pa, false, vb[t], (short)0, o[t], false, false);
        }
        // next iteration's sync(1) orders sPart/sP reuse -> no 3rd barrier
    }

    // ---- epilogue: normalize and store f32 -----------------------------
#pragma unroll
    for (int r = 0; r < 8; ++r) {
        const int row = q0 + r + 8 * hi;
        const float inv = 1.0f / sL[r + 8 * hi];
#pragma unroll
        for (int t = 0; t < 8; ++t) {
            const int n = wave * 128 + t * 16 + am;
            Out[(size_t)row * DIM + n] = o[t][r] * inv;
        }
    }
}

// ---------------------------------------------------------------------------
extern "C" void kernel_launch(void* const* d_in, const int* in_sizes, int n_in,
                              void* d_out, int out_size, void* d_ws, size_t ws_size,
                              hipStream_t stream) {
    const float* X  = (const float*)d_in[0];   // [4096,1024]
    const float* Wq = (const float*)d_in[1];   // [1024,1024]
    const float* Wk = (const float*)d_in[2];
    const float* Wv = (const float*)d_in[3];
    float* Out = (float*)d_out;                // [4096,1024]

    // Workspace carve-up (ushorts): Xbf | Wqbf | Wkbf | Wvbf | Qbf | Kbf | Vt
    unsigned short* Xbf  = (unsigned short*)d_ws;
    unsigned short* Wqbf = Xbf  + (size_t)SEQ * DIM;
    unsigned short* Wkbf = Wqbf + (size_t)DIM * DIM;
    unsigned short* Wvbf = Wkbf + (size_t)DIM * DIM;
    unsigned short* Qbf  = Wvbf + (size_t)DIM * DIM;
    unsigned short* Kbf  = Qbf  + (size_t)SEQ * DIM;
    unsigned short* Vt   = Kbf  + (size_t)SEQ * DIM;
    (void)in_sizes; (void)n_in; (void)out_size; (void)ws_size;

    // Kernel 0: one-shot conversions (bandwidth-bound, trivial VALU cost)
    cvt_f32_bf16<<<1024, 256, 0, stream>>>(X,  Xbf,  (SEQ * DIM) / 4);
    cvt_f32_bf16<<<512,  256, 0, stream>>>(Wq, Wqbf, (DIM * DIM) / 4);
    cvt_f32_bf16<<<512,  256, 0, stream>>>(Wk, Wkbf, (DIM * DIM) / 4);
    cvt_f32_bf16<<<512,  256, 0, stream>>>(Wv, Wvbf, (DIM * DIM) / 4);

    // Kernel 1: 256 M-tiles x {Q,K,V}; 8 waves cover all 1024 cols
    qkv_gemm_bf16<<<dim3(256, 3), 256, 0, stream>>>(Xbf, Wqbf, Wkbf, Wvbf,
                                                    Qbf, Kbf, Vt);

    // Kernel 2: one workgroup per 16 query rows
    flash_attn_bf16<<<dim3(SEQ / QBLK), 256, 0, stream>>>(Qbf, Kbf, Vt, Out);
}
